// LIFNeuron_10316511445287
// MI455X (gfx1250) — compile-verified, matrix operand order checked by reading, not run on hardware
//
#include <hip/hip_runtime.h>
#include <hip/hip_bf16.h>

typedef __attribute__((ext_vector_type(16))) _Float16 v16h;
typedef __attribute__((ext_vector_type(8)))  float    v8f;

#define K_TAPS 64
#define TSTEP  16   // timesteps per reduction chunk (WMMA M dim)
#define TBLK   128  // timesteps per block

__device__ __forceinline__ unsigned wave_ballot32(bool p) {
#if __has_builtin(__builtin_amdgcn_ballot_w32)
    return __builtin_amdgcn_ballot_w32(p);
#else
    return (unsigned)__ballot((int)p);
#endif
}

__device__ __forceinline__ float bcast_lane(float v, int srcLane) {
    return __int_as_float(__builtin_amdgcn_readlane(__float_as_int(v), srcLane));
}

// One block = all D=1024 channels of one batch row, for a 128-timestep slab.
// Thread d owns channel d. Per-thread exponential recurrence over time:
//   v[t] = tau*v[t-1] + x[t] - tau^64 * x[t-64]        (exact truncated conv)
// seeded at the slab start by a 64-tap warm-up window, so slabs are independent.
// Denominator (sum of spikes over 1024 channels) per timestep:
//   ballot+popcount per wave (32->1), then one v_wmma_f32_16x16x32_f16 with an
//   all-ones B matrix sums the 32 wave partials for 16 timesteps at once.
__global__ __launch_bounds__(1024, 1)
void LIFNeuron_kernel(const float* __restrict__ x,
                      const float* __restrict__ thr,
                      const float* __restrict__ tau_in,
                      const float* __restrict__ v0,
                      float* __restrict__ out,
                      int T, int D) {
    const int d    = threadIdx.x;          // channel, 0..1023
    const int lane = d & 31;
    const int wave = d >> 5;               // 0..31
    const int nslab = T / TBLK;
    const int b    = blockIdx.x / nslab;
    const int t0   = (blockIdx.x % nslab) * TBLK;
    const long long base = (long long)b * T * D + d;

    __shared__ _Float16 part[32 * TSTEP];  // [wave][timestep] spike popcounts
    if (threadIdx.x < 32 * TSTEP) part[threadIdx.x] = (_Float16)0.0f;

    // per-channel constants
    const float tc    = fminf(fmaxf(tau_in[d], 0.001f), 0.999f);
    const float lt    = __logf(tc);
    const float tau64 = __expf(64.0f * lt);
    const float th    = thr[d];

    // warm-up: v_prev = v_int[t0-1] = sum_{j=0..63} tc^j x[t0-1-j] + v0*tc^t0
    float vprev = v0[d] * __expf((float)t0 * lt);
    if (t0 >= K_TAPS) {
        float p = 1.0f;
        #pragma unroll
        for (int j = 0; j < K_TAPS; ++j) {
            vprev = fmaf(p, x[base + (long long)(t0 - 1 - j) * D], vprev);
            p *= tc;
        }
    }
    // (t0 == 0: v_int[-1] = v0, empty window)

    v16h bones;
    #pragma unroll
    for (int i = 0; i < 16; ++i) bones[i] = (_Float16)1.0f;

    __syncthreads();

    for (int tchunk = t0; tchunk < t0 + TBLK; tchunk += TSTEP) {
        unsigned smask = 0;
        float pcs = 0.0f;                  // this lane's slot of the popcount row
        #pragma unroll
        for (int i = 0; i < TSTEP; ++i) {
            const int t = tchunk + i;
            const float xt = x[base + (long long)t * D];
            const float xl = (t >= K_TAPS) ? x[base + (long long)(t - K_TAPS) * D]
                                           : 0.0f;
            float v = fmaf(tc, vprev, xt);
            v = fmaf(-tau64, xl, v);
            vprev = v;
            const bool sp = (v >= th);
            smask |= (sp ? 1u : 0u) << i;
            const unsigned bal = wave_ballot32(sp);
            const float pc = (float)__builtin_popcount(bal); // wave spikes @ t
            pcs = (lane == i) ? pc : pcs;                    // select, store once
        }
        if (lane < TSTEP) part[wave * TSTEP + lane] = (_Float16)pcs;
        __syncthreads();

        // A[m=timestep, k=wave] = part[k][m], packed per the documented
        // 16-bit 16x32 A layout; B = ones => D[m,*] = denom[timestep m].
        const int M  = lane & 15;
        const int hs = (lane >= 16) ? 8 : 0;
        v16h a;
        #pragma unroll
        for (int vr = 0; vr < 8; ++vr) {
            const int bk = ((vr < 4) ? 0 : 16) + hs + 2 * (vr & 3);
            a[2 * vr]     = part[bk * TSTEP + M];
            a[2 * vr + 1] = part[(bk + 1) * TSTEP + M];
        }
        v8f c = {};
        c = __builtin_amdgcn_wmma_f32_16x16x32_f16(
                false, a, false, bones, (short)0, c, false, false);
        __syncthreads();   // protect `part` before next chunk rewrites it

        // lanes 0-15 hold denom[0..7] in c[0..7]; lanes 16-31 hold denom[8..15].
        // Reciprocal first (8 v_rcp), then wave-uniform broadcast via readlane.
        float inv[8];
        #pragma unroll
        for (int j = 0; j < 8; ++j)
            inv[j] = __builtin_amdgcn_rcpf(fmaxf(c[j], 1.0f));

        #pragma unroll
        for (int m = 0; m < TSTEP; ++m) {
            const float r = bcast_lane(inv[m & 7], (m < 8) ? 0 : 16);
            const float o = ((smask >> m) & 1u) ? r : 0.0f;
            out[base + (long long)(tchunk + m) * D] = o;
        }
    }
}

extern "C" void kernel_launch(void* const* d_in, const int* in_sizes, int n_in,
                              void* d_out, int out_size, void* d_ws, size_t ws_size,
                              hipStream_t stream) {
    const float* x   = (const float*)d_in[0];
    const float* thr = (const float*)d_in[1];
    const float* tau = (const float*)d_in[2];
    const float* v0  = (const float*)d_in[3];
    float* out = (float*)d_out;

    const int D  = in_sizes[1];          // 1024
    const int T  = 4096;                 // reference shape
    const int BT = in_sizes[0] / D;      // B*T
    const int B  = BT / T;               // 8

    dim3 grid(B * (T / TBLK));           // 256 blocks
    dim3 block(D);                       // 1024 threads = 32 waves, 1 ch/thread
    LIFNeuron_kernel<<<grid, block, 0, stream>>>(x, thr, tau, v0, out, T, D);
}
